// XCA_mod_12223476924471
// MI455X (gfx1250) — matrix operand order repaired and checked by reading, not compiled, not run on hardware
//
#include <hip/hip_runtime.h>

// ---------------------------------------------------------------------------
// Types
// ---------------------------------------------------------------------------
typedef __attribute__((ext_vector_type(16))) __bf16        v16bf;
typedef __attribute__((ext_vector_type(8)))  float         v8f;
typedef __attribute__((ext_vector_type(4))) unsigned int   u32x4;
typedef __attribute__((ext_vector_type(4)))  float         f32x4;

union FragU {
    v16bf bf;
    u32x4 u[2];
};

// ---------------------------------------------------------------------------
// bf16 helpers (round-to-nearest-even)
// ---------------------------------------------------------------------------
__device__ __forceinline__ unsigned short f2bf(float x) {
    unsigned int u = __float_as_uint(x);
    unsigned int r = u + 0x7FFFu + ((u >> 16) & 1u);
    return (unsigned short)(r >> 16);
}
__device__ __forceinline__ float bf2f(unsigned short s) {
    return __uint_as_float(((unsigned int)s) << 16);
}

// ---------------------------------------------------------------------------
// CDNA5 async copy: global -> LDS, 16B per lane, tracked by ASYNCcnt
// (in-order completion => partial-count waits are safe).
// ---------------------------------------------------------------------------
__device__ __forceinline__ void async_load_b128(const unsigned short* lds_dst,
                                                const unsigned short* gsrc) {
    unsigned off = (unsigned)(unsigned long long)lds_dst;
    asm volatile("global_load_async_to_lds_b128 %0, %1, off"
                 :: "v"(off), "v"(gsrc)
                 : "memory");
}
#define WAIT_ASYNC(n) asm volatile("s_wait_asynccnt " #n ::: "memory")

// ---------------------------------------------------------------------------
// WMMA fragment helpers (layouts per CDNA5 ISA §7.12.2, wave32)
//   A 16x32 bf16: lane L: m = L&15, half = L>>4,
//     elems 0..7  -> K = half*8 + j ; elems 8..15 -> K = 16 + half*8 + (j-8)
//   B 32x16 bf16 from LDS laid out [col][K] (K contiguous):
//     lane L: col = L&15, same K order as A
// ---------------------------------------------------------------------------
__device__ __forceinline__ v16bf load_frag_a(const unsigned short* base, int ld) {
    const int lane = threadIdx.x & 31;
    const int m    = lane & 15;
    const int hl   = lane >> 4;
    const unsigned short* r = base + m * ld;
    FragU f;
    f.u[0] = *(const u32x4*)(r + hl * 8);
    f.u[1] = *(const u32x4*)(r + 16 + hl * 8);
    return f.bf;
}

__device__ __forceinline__ v16bf load_frag_b(const unsigned short* base, int ld) {
    const int lane = threadIdx.x & 31;
    const int n    = lane & 15;
    const int kh   = lane >> 4;
    const unsigned short* r = base + n * ld;
    FragU f;
    f.u[0] = *(const u32x4*)(r + kh * 8);
    f.u[1] = *(const u32x4*)(r + 16 + kh * 8);
    return f.bf;
}

__device__ __forceinline__ v8f wmma_bf16(v16bf a, v16bf b, v8f c) {
    // (neg_a, A, neg_b, B, c_mod, C, reuse_a, reuse_b)
    return __builtin_amdgcn_wmma_f32_16x16x32_bf16(false, a, false, b,
                                                   (short)0, c, false, false);
}

// transposed scatter of 8 bf16 (one 16B global load) into LDS column `col`
__device__ __forceinline__ void scatter_col(unsigned short* dst_row0, u32x4 v,
                                            int col, int ld) {
    const unsigned short* pv = (const unsigned short*)&v;
    #pragma unroll
    for (int i = 0; i < 8; ++i) dst_row0[i * ld + col] = pv[i];
}

// ---------------------------------------------------------------------------
// Problem constants
// ---------------------------------------------------------------------------
#define BATCH   32
#define CDIM    256
#define NDIM    3136       // 56*56
#define NHEADS  8
#define DHEAD   32
#define O3      768        // 3*C
#define LDSP    40         // padded LDS row stride (ushorts), 16B-aligned rows

// ---------------------------------------------------------------------------
// Kernel 1: fp32 -> bf16 convert of x  (25,690,112 elements, 4 per thread)
// ---------------------------------------------------------------------------
__global__ void __launch_bounds__(256) cvt_x_kernel(const float* __restrict__ x,
                                                    unsigned short* __restrict__ xb) {
    long i = (long)blockIdx.x * blockDim.x + threadIdx.x;
    long i4 = i * 4;
    const long total = (long)BATCH * CDIM * NDIM;
    if (i4 >= total) return;
    f32x4 v = *(const f32x4*)(x + i4);
    unsigned long long p =
        (unsigned long long)f2bf(v[0]) |
        ((unsigned long long)f2bf(v[1]) << 16) |
        ((unsigned long long)f2bf(v[2]) << 32) |
        ((unsigned long long)f2bf(v[3]) << 48);
    *(unsigned long long*)(xb + i4) = p;
}

// ---------------------------------------------------------------------------
// Kernel 2: weight standardization + bf16 convert.
// Rows 0..767: qkv_w/qkv_gain;  rows 768..1023: proj_w/proj_gain.  K = 256.
// ---------------------------------------------------------------------------
__global__ void __launch_bounds__(256) std_weight_kernel(
        const float* __restrict__ qkv_w, const float* __restrict__ qkv_gain,
        const float* __restrict__ proj_w, const float* __restrict__ proj_gain,
        unsigned short* __restrict__ wq_bf, unsigned short* __restrict__ wp_bf) {
    const int row = blockIdx.x;
    const int t   = threadIdx.x;
    const float* w;
    const float* g;
    unsigned short* o;
    int r;
    if (row < O3) { r = row;       w = qkv_w  + (long)r * CDIM; g = qkv_gain;  o = wq_bf + (long)r * CDIM; }
    else          { r = row - O3;  w = proj_w + (long)r * CDIM; g = proj_gain; o = wp_bf + (long)r * CDIM; }

    float x = w[t];
    __shared__ float s1[256];
    __shared__ float s2[256];
    s1[t] = x; s2[t] = x * x;
    __syncthreads();
    for (int s = 128; s > 0; s >>= 1) {
        if (t < s) { s1[t] += s1[t + s]; s2[t] += s2[t + s]; }
        __syncthreads();
    }
    float mean = s1[0] * (1.0f / 256.0f);
    float var  = s2[0] * (1.0f / 256.0f) - mean * mean;
    float sc   = g[r] * 0.0625f * rsqrtf(var + 1e-6f);   // 0.0625 = 1/sqrt(256)
    o[t] = f2bf((x - mean) * sc);
}

// ---------------------------------------------------------------------------
// Kernel 3/6: bf16 GEMM  out[b] = A(MxK) * B[b](KxN) (+bias, +residual)
// Block tile 64x64, K-step 32, 8 waves (4 in M x 2 in N), 2 WMMA/wave/step.
// Depth-3 software pipeline: A tile via async-to-LDS (wait asynccnt<=1 keeps
// the newest load in flight); B tile global->reg one step ahead, reg->LDS
// scatter the step after. One barrier per K-step.
// MODE 0: out bf16 = acc + bias[m]            (QKV GEMM)
// MODE 1: out f32  = acc + bias[m] + res      (proj GEMM + residual)
// ---------------------------------------------------------------------------
template<int MODE>
__global__ void __launch_bounds__(256) gemm_kernel(
        const unsigned short* __restrict__ A,    // M x K bf16 (weights)
        const unsigned short* __restrict__ Bm,   // BATCH x K x N bf16
        const float* __restrict__ bias,          // M
        unsigned short* __restrict__ outb,       // MODE 0
        float* __restrict__ outf,                // MODE 1
        const float* __restrict__ res,           // MODE 1 residual (same layout)
        int M, int N, int K) {
    const int n0 = blockIdx.x * 64;
    const int m0 = blockIdx.y * 64;
    const int b  = blockIdx.z;
    const unsigned short* Bb = Bm + (long)b * K * N;

    __shared__ unsigned short ldsA[3][64 * LDSP];   // [m][k]
    __shared__ unsigned short ldsB[3][64 * LDSP];   // [n][k]  (staged transposed)

    const int t    = threadIdx.x;
    const int lane = t & 31;
    const int w    = t >> 5;
    const int wm   = w & 3;       // 0..3 -> 16-row slab
    const int wn   = w >> 2;      // 0..1 -> 32-col slab

    // per-thread staging coordinates
    const int am  = t >> 2, ac8 = (t & 3) * 8;     // A tile: 64 x 32
    const int bci = t >> 3, bnj = (t & 7) * 8;     // B tile: 32(k) x 64(n)
    const unsigned short* gA = A  + (long)(m0 + am) * K + ac8;
    const unsigned short* gB = Bb + (long)bci * N + n0 + bnj;

    const int ksteps = K >> 5;                     // 8 for K=256

    // ---- prologue: async A(0),A(1); B(0) staged; B(1) held in regs ----
    async_load_b128(&ldsA[0][am * LDSP + ac8], gA);
    if (ksteps > 1)
        async_load_b128(&ldsA[1][am * LDSP + ac8], gA + 32);
    {
        u32x4 v0 = *(const u32x4*)gB;
        scatter_col(&ldsB[0][bnj * LDSP], v0, bci, LDSP);
    }
    u32x4 bvN = {};
    if (ksteps > 1) bvN = *(const u32x4*)(gB + (long)32 * N);

    v8f acc0 = {};
    v8f acc1 = {};

    for (int ks = 0; ks < ksteps; ++ks) {
        const int cur = ks % 3;
        // A(ks) must have landed; A(ks+1) may remain in flight.
        if (ks + 1 < ksteps) { WAIT_ASYNC(1); } else { WAIT_ASYNC(0); }
        __syncthreads();

        if (ks + 2 < ksteps)
            async_load_b128(&ldsA[(ks + 2) % 3][am * LDSP + ac8],
                            gA + (ks + 2) * 32);
        if (ks + 1 < ksteps)
            scatter_col(&ldsB[(ks + 1) % 3][bnj * LDSP], bvN, bci, LDSP);
        if (ks + 2 < ksteps)
            bvN = *(const u32x4*)(gB + (long)(ks + 2) * 32 * N);

        v16bf a  = load_frag_a(&ldsA[cur][wm * 16 * LDSP], LDSP);
        v16bf b0 = load_frag_b(&ldsB[cur][(wn * 32) * LDSP], LDSP);
        v16bf b1 = load_frag_b(&ldsB[cur][(wn * 32 + 16) * LDSP], LDSP);
        acc0 = wmma_bf16(a, b0, acc0);
        acc1 = wmma_bf16(a, b1, acc1);
    }

    // Epilogue: D layout -> lane col = lane&15, VGPR r -> row r + (lane>>4)*8
    #pragma unroll
    for (int r = 0; r < 8; ++r) {
        const int mg  = m0 + wm * 16 + r + (lane >> 4) * 8;
        const int nl  = lane & 15;
        const float bv = bias[mg];
        const long o0 = (long)b * M * N + (long)mg * N + n0 + wn * 32 + nl;
        if (MODE == 0) {
            outb[o0]      = f2bf(acc0[r] + bv);
            outb[o0 + 16] = f2bf(acc1[r] + bv);
        } else {
            outf[o0]      = res[o0]      + acc0[r] + bv;
            outf[o0 + 16] = res[o0 + 16] + acc1[r] + bv;
        }
    }
}

// ---------------------------------------------------------------------------
// Kernel 4: inverse L2 norms of q,k rows. rid = b*512 + o, o in [0,512).
// ---------------------------------------------------------------------------
__global__ void __launch_bounds__(256) rownorm_kernel(
        const unsigned short* __restrict__ qkv, float* __restrict__ rnorm) {
    const int rid = blockIdx.x;               // 0..16383
    const int b   = rid >> 9;
    const int o   = rid & 511;
    const unsigned short* p = qkv + ((long)b * O3 + o) * NDIM;
    float ss = 0.0f;
    for (int i = threadIdx.x; i < NDIM; i += 256) {
        float f = bf2f(p[i]);
        ss += f * f;
    }
    __shared__ float s[256];
    s[threadIdx.x] = ss;
    __syncthreads();
    for (int st = 128; st > 0; st >>= 1) {
        if (threadIdx.x < st) s[threadIdx.x] += s[threadIdx.x + st];
        __syncthreads();
    }
    if (threadIdx.x == 0) rnorm[rid] = 1.0f / fmaxf(sqrtf(s[0]), 1e-12f);
}

// ---------------------------------------------------------------------------
// Kernel 5: fused XCA attention per (b,h). 128 threads = 4 waves.
//   Phase 1: S = q * k^T  (32x32, K=3136) -- depth-3 async pipeline,
//            wait asynccnt<=2 keeps the newest Q/K pair in flight
//   Phase 2: scale by inv-norms * temperature, softmax over e (32)
//   Phase 3: out = P * V  (32 x 3136, K=32) -- depth-3 reg-carried V transpose
// ---------------------------------------------------------------------------
__global__ void __launch_bounds__(128) attn_kernel(
        const unsigned short* __restrict__ qkv,
        const float* __restrict__ rnorm,
        const float* __restrict__ temperature,
        unsigned short* __restrict__ attnout) {
    const int bh = blockIdx.x;
    const int b  = bh >> 3;
    const int h  = bh & 7;

    const long baseQ = ((long)b * O3 + h * DHEAD) * NDIM;
    const long baseK = baseQ + (long)CDIM * NDIM;
    const long baseV = baseQ + 2L * CDIM * NDIM;

    __shared__ unsigned short ldsQ[3][32 * LDSP];   // [d][n]  (A layout)
    __shared__ unsigned short ldsK[3][32 * LDSP];   // [e][n]  (= B [col=e][K=n])
    __shared__ unsigned short ldsVT[3][32 * LDSP];  // [n][e]  (= B [col=n][K=e])
    __shared__ unsigned short ldsP[32 * LDSP];      // softmax probs, A layout
    __shared__ float ldsS[32 * 32];
    __shared__ float ldsQR[32];
    __shared__ float ldsKR[32];

    const int t    = threadIdx.x;
    const int lane = t & 31;
    const int w    = t >> 5;     // 0..3
    const int md   = w & 1;      // row half of 32
    const int ne   = w >> 1;     // col half of 32
    const int NCH  = NDIM / 32;  // 98

    // staging coordinates: 32 rows x 32 cols, 16B per thread
    const int lrow = t >> 2;           // 0..31
    const int lc8  = (t & 3) * 8;      // 0,8,16,24
    const unsigned short* qsrc0 = qkv + baseQ + (long)lrow * NDIM + lc8;
    const unsigned short* ksrc0 = qkv + baseK + (long)lrow * NDIM + lc8;
    const unsigned short* vsrc0 = qkv + baseV + (long)lrow * NDIM + lc8;

    if (t < 32) {
        ldsQR[t] = rnorm[b * 512 + h * DHEAD + t];
        ldsKR[t] = rnorm[b * 512 + 256 + h * DHEAD + t];
    }

    // ---- Phase 1: S = q k^T, depth-3 async pipeline ----
    async_load_b128(&ldsQ[0][lrow * LDSP + lc8], qsrc0);
    async_load_b128(&ldsK[0][lrow * LDSP + lc8], ksrc0);
    async_load_b128(&ldsQ[1][lrow * LDSP + lc8], qsrc0 + 32);
    async_load_b128(&ldsK[1][lrow * LDSP + lc8], ksrc0 + 32);

    v8f accS = {};
    for (int nc = 0; nc < NCH; ++nc) {
        const int cur = nc % 3;
        // pair(nc) must have landed; pair(nc+1) may remain in flight.
        if (nc + 1 < NCH) { WAIT_ASYNC(2); } else { WAIT_ASYNC(0); }
        __syncthreads();
        if (nc + 2 < NCH) {
            async_load_b128(&ldsQ[(nc + 2) % 3][lrow * LDSP + lc8],
                            qsrc0 + (nc + 2) * 32);
            async_load_b128(&ldsK[(nc + 2) % 3][lrow * LDSP + lc8],
                            ksrc0 + (nc + 2) * 32);
        }
        v16bf a  = load_frag_a(&ldsQ[cur][md * 16 * LDSP], LDSP);
        v16bf bb = load_frag_b(&ldsK[cur][ne * 16 * LDSP], LDSP);
        accS = wmma_bf16(a, bb, accS);
    }
    __syncthreads();

    // spill S to LDS
    #pragma unroll
    for (int r = 0; r < 8; ++r) {
        const int m = md * 16 + r + (lane >> 4) * 8;
        const int n = ne * 16 + (lane & 15);
        ldsS[m * 32 + n] = accS[r];
    }
    __syncthreads();

    // ---- Phase 2: scaled softmax over e (one row per thread 0..31) ----
    if (t < 32) {
        const int d    = t;
        const float qr = ldsQR[d];
        const float tp = temperature[h];
        float sv[32];
        float mx = -1e30f;
        #pragma unroll
        for (int e = 0; e < 32; ++e) {
            float s = ldsS[d * 32 + e] * qr * ldsKR[e] * tp;
            sv[e] = s;
            mx = fmaxf(mx, s);
        }
        float sum = 0.0f;
        #pragma unroll
        for (int e = 0; e < 32; ++e) { float p = __expf(sv[e] - mx); sv[e] = p; sum += p; }
        const float inv = 1.0f / sum;
        #pragma unroll
        for (int e = 0; e < 32; ++e) ldsP[d * LDSP + e] = f2bf(sv[e] * inv);
    }
    __syncthreads();

    // ---- Phase 3: out = P * V, depth-3 reg-carried transpose pipeline ----
    const long baseO = ((long)b * CDIM + h * DHEAD) * NDIM;
    {
        u32x4 v0 = *(const u32x4*)vsrc0;
        scatter_col(&ldsVT[0][lc8 * LDSP], v0, lrow, LDSP);
    }
    u32x4 vvN = {};
    if (NCH > 1) vvN = *(const u32x4*)(vsrc0 + 32);

    for (int nc = 0; nc < NCH; ++nc) {
        const int cur = nc % 3;
        const int n0  = nc * 32;
        __syncthreads();
        if (nc + 1 < NCH)
            scatter_col(&ldsVT[(nc + 1) % 3][lc8 * LDSP], vvN, lrow, LDSP);
        if (nc + 2 < NCH)
            vvN = *(const u32x4*)(vsrc0 + (nc + 2) * 32);

        v16bf a  = load_frag_a(&ldsP[md * 16 * LDSP], LDSP);
        v16bf bb = load_frag_b(&ldsVT[cur][ne * 16 * LDSP], LDSP);
        v8f c = {};
        v8f d = wmma_bf16(a, bb, c);
        #pragma unroll
        for (int r = 0; r < 8; ++r) {
            const int m = md * 16 + r + (lane >> 4) * 8;
            const int n = ne * 16 + (lane & 15);
            attnout[baseO + (long)m * NDIM + n0 + n] = f2bf(d[r]);
        }
    }
}

// ---------------------------------------------------------------------------
// Host launcher
// ---------------------------------------------------------------------------
extern "C" void kernel_launch(void* const* d_in, const int* in_sizes, int n_in,
                              void* d_out, int out_size, void* d_ws, size_t ws_size,
                              hipStream_t stream) {
    (void)in_sizes; (void)n_in; (void)out_size; (void)ws_size;

    const float* x           = (const float*)d_in[0];
    const float* qkv_w       = (const float*)d_in[1];
    const float* qkv_b       = (const float*)d_in[2];
    const float* qkv_gain    = (const float*)d_in[3];
    const float* proj_w      = (const float*)d_in[4];
    const float* proj_b      = (const float*)d_in[5];
    const float* proj_gain   = (const float*)d_in[6];
    const float* temperature = (const float*)d_in[7];
    float* out               = (float*)d_out;

    char* ws = (char*)d_ws;
    size_t off = 0;
    auto alloc = [&](size_t bytes) -> void* {
        off = (off + 255) & ~(size_t)255;
        void* p = ws + off;
        off += bytes;
        return p;
    };

    unsigned short* wq_bf  = (unsigned short*)alloc((size_t)O3 * CDIM * 2);
    unsigned short* wp_bf  = (unsigned short*)alloc((size_t)CDIM * CDIM * 2);
    unsigned short* x_bf   = (unsigned short*)alloc((size_t)BATCH * CDIM * NDIM * 2);
    unsigned short* qkv_bf = (unsigned short*)alloc((size_t)BATCH * O3 * NDIM * 2);
    float*          rnorm  = (float*)alloc((size_t)BATCH * 512 * 4);
    unsigned short* ao_bf  = (unsigned short*)alloc((size_t)BATCH * CDIM * NDIM * 2);

    // 1) x -> bf16   (25,690,112 / (256*4) = 25088 blocks exactly)
    cvt_x_kernel<<<25088, 256, 0, stream>>>(x, x_bf);

    // 2) standardized weights -> bf16 (768 qkv rows + 256 proj rows)
    std_weight_kernel<<<O3 + CDIM, 256, 0, stream>>>(
        qkv_w, qkv_gain, proj_w, proj_gain, wq_bf, wp_bf);

    // 3) QKV GEMM: (768x256) x (256x3136) per batch, + bias, bf16 out
    gemm_kernel<0><<<dim3(NDIM / 64, O3 / 64, BATCH), 256, 0, stream>>>(
        wq_bf, x_bf, qkv_b, qkv_bf, nullptr, nullptr, O3, NDIM, CDIM);

    // 4) inverse L2 norms for q,k rows
    rownorm_kernel<<<BATCH * 512, 256, 0, stream>>>(qkv_bf, rnorm);

    // 5) fused XCA attention, one block per (b,h)
    attn_kernel<<<BATCH * NHEADS, 128, 0, stream>>>(qkv_bf, rnorm, temperature, ao_bf);

    // 6) proj GEMM + bias + residual, fp32 out
    gemm_kernel<1><<<dim3(NDIM / 64, CDIM / 64, BATCH), 256, 0, stream>>>(
        wp_bf, ao_bf, proj_b, nullptr, out, x, CDIM, NDIM, CDIM);
}
